// SingleGPUMoETorchFFN_83442624627174
// MI455X (gfx1250) — compile-verified, matrix operand order checked by reading, not run on hardware
//
#include <hip/hip_runtime.h>

// ---------------------------------------------------------------------------
// MoE SwiGLU FFN, top-2 of 8 experts, for gfx1250 (MI455X).
// x:[T=2048,D=2048] f32, gate_w:[8,D], w1/w3:[8,H=5632,D], w2:[8,D,H]
//
// Routed math = 283 GFLOP; weights = 1.1 GB fp32 (~48us compulsory @23.3TB/s);
// bf16 WMMA compute ~113us.  BM=256 blocking -> weights re-read only ~2x per
// expert (~95us HBM), balanced vs the matrix pipes.  h (46 MB bf16) and a
// gathered bf16 x (16 MB) live in d_ws; cross-N-block A re-reads hit the
// 192 MB L2.  GEMM K-loops are double-buffered through LDS: global loads for
// panel k+1 issue before the 16 WMMAs of panel k, stores drain afterwards.
// ---------------------------------------------------------------------------

#define T_TOK 2048
#define DDIM  2048
#define EEXP  8
#define HDIM  5632
#define MT    16

#define BM 256
#define BN 64
#define KB 32
#define LP 40                 // LDS row pitch (bf16 elems): 80B rows, 16B aligned
#define ABUF (BM * LP)        // elements per A buffer
#define BBUF (BN * LP)        // elements per B buffer

#define MBH (T_TOK / BM)      // 8
#define NBH (HDIM / BN)       // 88
#define NBY (DDIM / BN)       // 32
#define NKH (DDIM / KB)       // 64
#define NKY (HDIM / KB)       // 176

// Workspace layout
#define WS_COUNTS 0
#define WS_OFFS   64
#define WS_TLIST  1024
#define WS_WLIST  (1024 + EEXP * T_TOK * 4)
#define WS_XG     (WS_WLIST + EEXP * T_TOK * 4)
#define XG_BYTES  ((size_t)2 * T_TOK * DDIM * 2)
#define WS_HBUF   (WS_XG + XG_BYTES)
#define HB_BYTES  ((size_t)2 * T_TOK * HDIM * 2)
#define WS_REQ    (WS_HBUF + HB_BYTES)

typedef __attribute__((ext_vector_type(16))) __bf16 v16bf;
typedef __attribute__((ext_vector_type(8)))  __bf16 v8bf;
typedef __attribute__((ext_vector_type(8)))  float  v8f;

static __device__ inline v16bf cat8(v8bf lo, v8bf hi) {
  return __builtin_shufflevector(lo, hi, 0,1,2,3,4,5,6,7,8,9,10,11,12,13,14,15);
}
static __device__ inline v16bf pack_lds(const __bf16* p0, const __bf16* p1) {
  return cat8(*(const v8bf*)p0, *(const v8bf*)p1);
}
static __device__ inline v16bf pack_f32(const float* p0, const float* p1) {
  return cat8(__builtin_convertvector(*(const v8f*)p0, v8bf),
              __builtin_convertvector(*(const v8f*)p1, v8bf));
}
static __device__ inline float silu_fast(float g) {
  return g * __builtin_amdgcn_rcpf(1.0f + __expf(-g));
}
#define WMMA_BF16(a, b, c) \
  __builtin_amdgcn_wmma_f32_16x16x32_bf16(false, (a), false, (b), (short)0, (c), false, false)

// ---------------------------------------------------------------------------
__global__ void moe_init(float* __restrict__ out, int* __restrict__ counts) {
  size_t i = (size_t)blockIdx.x * blockDim.x + threadIdx.x;
  const size_t n = (size_t)T_TOK * DDIM;
  for (size_t k = i; k < n; k += (size_t)gridDim.x * blockDim.x) out[k] = 0.0f;
  if (i < EEXP) counts[i] = 0;
}

// ---------------------------------------------------------------------------
__global__ __launch_bounds__(256) void moe_gate(const float* __restrict__ x,
                                                const float* __restrict__ gwm,
                                                int* __restrict__ counts,
                                                int* __restrict__ tlist,
                                                float* __restrict__ wlist) {
  const int lane = threadIdx.x & 31;
  const int t = blockIdx.x * 8 + (threadIdx.x >> 5);
  float s[EEXP];
#pragma unroll
  for (int e = 0; e < EEXP; ++e) s[e] = 0.0f;
  const float* xr = x + (size_t)t * DDIM;
  for (int d = lane; d < DDIM; d += 32) {
    const float xv = xr[d];
#pragma unroll
    for (int e = 0; e < EEXP; ++e) s[e] += xv * gwm[e * DDIM + d];
  }
#pragma unroll
  for (int e = 0; e < EEXP; ++e) {
#pragma unroll
    for (int off = 16; off > 0; off >>= 1) s[e] += __shfl_xor(s[e], off, 32);
  }
  if (lane == 0) {
    int e0 = 0;
#pragma unroll
    for (int e = 1; e < EEXP; ++e) if (s[e] > s[e0]) e0 = e;
    int e1 = (e0 == 0) ? 1 : 0;
#pragma unroll
    for (int e = 0; e < EEXP; ++e) if (e != e0 && s[e] > s[e1]) e1 = e;
    const float z1 = __expf(s[e1] - s[e0]);
    const float inv = __builtin_amdgcn_rcpf(1.0f + z1);
    const int p0 = atomicAdd(&counts[e0], 1);
    tlist[e0 * T_TOK + p0] = t;  wlist[e0 * T_TOK + p0] = inv;
    const int p1 = atomicAdd(&counts[e1], 1);
    tlist[e1 * T_TOK + p1] = t;  wlist[e1 * T_TOK + p1] = z1 * inv;
  }
}

// ---------------------------------------------------------------------------
__global__ void moe_offsets(const int* __restrict__ counts, int* __restrict__ offs) {
  if (threadIdx.x == 0 && blockIdx.x == 0) {
    int a = 0;
#pragma unroll
    for (int e = 0; e < EEXP; ++e) { offs[e] = a; a += counts[e]; }
    offs[EEXP] = a;
  }
}

// ---------------------------------------------------------------------------
__global__ __launch_bounds__(256) void moe_gather(const float* __restrict__ x,
                                                  const int* __restrict__ counts,
                                                  const int* __restrict__ offs,
                                                  const int* __restrict__ tlist,
                                                  __bf16* __restrict__ xg) {
  const int e  = blockIdx.x >> 8;
  const int sb = (blockIdx.x & 255) * 8 + (threadIdx.x >> 5);
  if (sb >= counts[e]) return;
  const int t = tlist[e * T_TOK + sb];
  const float4* src = (const float4*)(x + (size_t)t * DDIM);
  __bf16* dst = xg + (size_t)(offs[e] + sb) * DDIM;
  for (int c = threadIdx.x & 31; c < DDIM / 4; c += 32) {
    const float4 v = src[c];
    __bf16* d = dst + c * 4;
    d[0] = (__bf16)v.x; d[1] = (__bf16)v.y; d[2] = (__bf16)v.z; d[3] = (__bf16)v.w;
  }
}

// ---------------------------------------------------------------------------
// GEMM-A: h = silu(xg w1^T) * (xg w3^T) -> hbuf (bf16).
// BM=256 x BN=64, double-buffered LDS, 1 barrier / K-step, 16 WMMA / K-step.
// ---------------------------------------------------------------------------
__global__ __launch_bounds__(256) void moe_gemm_h(const __bf16* __restrict__ xg,
                                                  const float* __restrict__ w1,
                                                  const float* __restrict__ w3,
                                                  const int* __restrict__ counts,
                                                  const int* __restrict__ offs,
                                                  __bf16* __restrict__ hbuf) {
  __shared__ __align__(16) __bf16 As[2 * ABUF];    // 40 KB
  __shared__ __align__(16) __bf16 Bs1[2 * BBUF];   // 10 KB
  __shared__ __align__(16) __bf16 Bs3[2 * BBUF];   // 10 KB

  const int e   = blockIdx.x / (MBH * NBH);
  const int rem = blockIdx.x % (MBH * NBH);
  const int mb  = rem / NBH;
  const int nb  = rem % NBH;
  const int cnt = counts[e];
  if (mb * BM >= cnt) return;
  const int oe = offs[e];

  const int tid = threadIdx.x, lane = tid & 31, wv = tid >> 5;
  const int hi = lane >> 4, ln = lane & 15;

  const float* w1e = w1 + (size_t)e * HDIM * DDIM + (size_t)nb * BN * DDIM;
  const float* w3e = w3 + (size_t)e * HDIM * DDIM + (size_t)nb * BN * DDIM;

  // K-invariant staging geometry: 4 A-chunks + 1 B-chunk (x2 matrices) / thread
  const int kcA = (tid & 3) * 8;
  const __bf16* srcA[4];
  int dstA[4];
#pragma unroll
  for (int i = 0; i < 4; ++i) {
    const int row  = (tid >> 2) + i * 64;
    const int slot = mb * BM + row;
    const int cs   = slot < cnt ? slot : cnt - 1;
    srcA[i] = xg + (size_t)(oe + cs) * DDIM + kcA;
    dstA[i] = row * LP + kcA;
  }
  const int rowB = tid >> 2, kcB = (tid & 3) * 8;
  const float* srcB1 = w1e + (size_t)rowB * DDIM + kcB;
  const float* srcB3 = w3e + (size_t)rowB * DDIM + kcB;
  const int dstB = rowB * LP + kcB;

  v8f acc1[2][4], acc3[2][4];
#pragma unroll
  for (int i = 0; i < 2; ++i)
#pragma unroll
    for (int j = 0; j < 4; ++j) { acc1[i][j] = (v8f){}; acc3[i][j] = (v8f){}; }

  // prologue: stage panel 0 into buffer 0
  {
    v8bf a[4];
#pragma unroll
    for (int i = 0; i < 4; ++i) a[i] = *(const v8bf*)srcA[i];
    const v8f b1v = *(const v8f*)srcB1;
    const v8f b3v = *(const v8f*)srcB3;
#pragma unroll
    for (int i = 0; i < 4; ++i) *(v8bf*)(As + dstA[i]) = a[i];
    *(v8bf*)(Bs1 + dstB) = __builtin_convertvector(b1v, v8bf);
    *(v8bf*)(Bs3 + dstB) = __builtin_convertvector(b3v, v8bf);
  }

  for (int k = 0; k < NKH; ++k) {
    __syncthreads();
    const int cur = k & 1, nxt = cur ^ 1;
    const bool have = (k + 1) < NKH;

    v8bf an[4]; v8f b1n, b3n;
    if (have) {                               // issue next-panel loads first
      const int k0n = (k + 1) * KB;
#pragma unroll
      for (int i = 0; i < 4; ++i) an[i] = *(const v8bf*)(srcA[i] + k0n);
      b1n = *(const v8f*)(srcB1 + k0n);
      b3n = *(const v8f*)(srcB3 + k0n);
      __builtin_prefetch(srcB1 + k0n + KB, 0, 1);
      __builtin_prefetch(srcB3 + k0n + KB, 0, 1);
    }

    // compute on current buffer (latency of the loads above hides under WMMA)
    const __bf16* Ac  = As  + cur * ABUF;
    const __bf16* B1c = Bs1 + cur * BBUF;
    const __bf16* B3c = Bs3 + cur * BBUF;
    v16bf af[2];
#pragma unroll
    for (int i = 0; i < 2; ++i) {
      const __bf16* ar = Ac + ((wv * 2 + i) * 16 + ln) * LP;
      af[i] = pack_lds(ar + hi * 8, ar + 16 + hi * 8);
    }
#pragma unroll
    for (int j = 0; j < 4; ++j) {
      const __bf16* b1r = B1c + (j * 16 + ln) * LP;
      const __bf16* b3r = B3c + (j * 16 + ln) * LP;
      const v16bf b1 = pack_lds(b1r + hi * 8, b1r + 16 + hi * 8);
      const v16bf b3 = pack_lds(b3r + hi * 8, b3r + 16 + hi * 8);
#pragma unroll
      for (int i = 0; i < 2; ++i) {
        acc1[i][j] = WMMA_BF16(af[i], b1, acc1[i][j]);
        acc3[i][j] = WMMA_BF16(af[i], b3, acc3[i][j]);
      }
    }

    if (have) {                               // drain into the other buffer
      __bf16* An = As + nxt * ABUF;
#pragma unroll
      for (int i = 0; i < 4; ++i) *(v8bf*)(An + dstA[i]) = an[i];
      *(v8bf*)(Bs1 + nxt * BBUF + dstB) = __builtin_convertvector(b1n, v8bf);
      *(v8bf*)(Bs3 + nxt * BBUF + dstB) = __builtin_convertvector(b3n, v8bf);
    }
  }

  // epilogue: silu(u)*v -> hbuf
#pragma unroll
  for (int i = 0; i < 2; ++i) {
#pragma unroll
    for (int j = 0; j < 4; ++j) {
#pragma unroll
      for (int r = 0; r < 8; ++r) {
        const int slot = mb * BM + (wv * 2 + i) * 16 + r + hi * 8;
        if (slot < cnt) {
          const float hval = silu_fast(acc1[i][j][r]) * acc3[i][j][r];
          hbuf[(size_t)(oe + slot) * HDIM + nb * BN + j * 16 + ln] = (__bf16)hval;
        }
      }
    }
  }
}

// ---------------------------------------------------------------------------
// GEMM-B: y[tok] += gate * (h w2^T); atomic top-2 combine.
// ---------------------------------------------------------------------------
__global__ __launch_bounds__(256) void moe_gemm_y(const __bf16* __restrict__ hbuf,
                                                  const float* __restrict__ w2,
                                                  const int* __restrict__ counts,
                                                  const int* __restrict__ offs,
                                                  const int* __restrict__ tlist,
                                                  const float* __restrict__ wlist,
                                                  float* __restrict__ out) {
  __shared__ __align__(16) __bf16 As[2 * ABUF];    // 40 KB
  __shared__ __align__(16) __bf16 Bs[2 * BBUF];    // 10 KB
  __shared__ int   tokr[BM];
  __shared__ float gwr[BM];

  const int e   = blockIdx.x / (MBH * NBY);
  const int rem = blockIdx.x % (MBH * NBY);
  const int mb  = rem / NBY;
  const int nb  = rem % NBY;
  const int cnt = counts[e];
  if (mb * BM >= cnt) return;
  const int oe = offs[e];

  const int tid = threadIdx.x, lane = tid & 31, wv = tid >> 5;
  const int hi = lane >> 4, ln = lane & 15;

  {
    const int slot = mb * BM + tid;
    if (slot < cnt) { tokr[tid] = tlist[e * T_TOK + slot]; gwr[tid] = wlist[e * T_TOK + slot]; }
    else            { tokr[tid] = 0;                        gwr[tid] = 0.0f; }
  } // first K-step barrier publishes these

  const float* w2e = w2 + (size_t)e * DDIM * HDIM + (size_t)nb * BN * HDIM;

  const int kcA = (tid & 3) * 8;
  const __bf16* srcA[4];
  int dstA[4];
#pragma unroll
  for (int i = 0; i < 4; ++i) {
    const int row  = (tid >> 2) + i * 64;
    const int slot = mb * BM + row;
    const int cs   = slot < cnt ? slot : cnt - 1;
    srcA[i] = hbuf + (size_t)(oe + cs) * HDIM + kcA;
    dstA[i] = row * LP + kcA;
  }
  const int rowB = tid >> 2, kcB = (tid & 3) * 8;
  const float* srcB = w2e + (size_t)rowB * HDIM + kcB;
  const int dstB = rowB * LP + kcB;

  v8f acc[2][4];
#pragma unroll
  for (int i = 0; i < 2; ++i)
#pragma unroll
    for (int j = 0; j < 4; ++j) acc[i][j] = (v8f){};

  {
    v8bf a[4];
#pragma unroll
    for (int i = 0; i < 4; ++i) a[i] = *(const v8bf*)srcA[i];
    const v8f bv = *(const v8f*)srcB;
#pragma unroll
    for (int i = 0; i < 4; ++i) *(v8bf*)(As + dstA[i]) = a[i];
    *(v8bf*)(Bs + dstB) = __builtin_convertvector(bv, v8bf);
  }

  for (int k = 0; k < NKY; ++k) {
    __syncthreads();
    const int cur = k & 1, nxt = cur ^ 1;
    const bool have = (k + 1) < NKY;

    v8bf an[4]; v8f bn;
    if (have) {
      const int k0n = (k + 1) * KB;
#pragma unroll
      for (int i = 0; i < 4; ++i) an[i] = *(const v8bf*)(srcA[i] + k0n);
      bn = *(const v8f*)(srcB + k0n);
      __builtin_prefetch(srcB + k0n + KB, 0, 1);
    }

    const __bf16* Ac = As + cur * ABUF;
    const __bf16* Bc = Bs + cur * BBUF;
    v16bf af[2];
#pragma unroll
    for (int i = 0; i < 2; ++i) {
      const __bf16* ar = Ac + ((wv * 2 + i) * 16 + ln) * LP;
      af[i] = pack_lds(ar + hi * 8, ar + 16 + hi * 8);
    }
#pragma unroll
    for (int j = 0; j < 4; ++j) {
      const __bf16* br = Bc + (j * 16 + ln) * LP;
      const v16bf b = pack_lds(br + hi * 8, br + 16 + hi * 8);
#pragma unroll
      for (int i = 0; i < 2; ++i) acc[i][j] = WMMA_BF16(af[i], b, acc[i][j]);
    }

    if (have) {
      __bf16* An = As + nxt * ABUF;
#pragma unroll
      for (int i = 0; i < 4; ++i) *(v8bf*)(An + dstA[i]) = an[i];
      *(v8bf*)(Bs + nxt * BBUF + dstB) = __builtin_convertvector(bn, v8bf);
    }
  }

#pragma unroll
  for (int i = 0; i < 2; ++i) {
#pragma unroll
    for (int j = 0; j < 4; ++j) {
#pragma unroll
      for (int r = 0; r < 8; ++r) {
        const int sl  = (wv * 2 + i) * 16 + r + hi * 8;   // gwr==0 for padded rows
        const int col = nb * BN + j * 16 + ln;
        atomicAdd(out + (size_t)tokr[sl] * DDIM + col, gwr[sl] * acc[i][j][r]);
      }
    }
  }
}

// ---------------------------------------------------------------------------
// Fallback (small d_ws): fused kernel, h in LDS, M=16 tiles.
// ---------------------------------------------------------------------------
#define HCHUNK 2816
__global__ __launch_bounds__(256) void moe_ffn_fused(const float* __restrict__ x,
                                                     const float* __restrict__ w1,
                                                     const float* __restrict__ w2,
                                                     const float* __restrict__ w3,
                                                     const int* __restrict__ counts,
                                                     const int* __restrict__ tlist,
                                                     const float* __restrict__ wlist,
                                                     float* __restrict__ out) {
  __shared__ __align__(16) __bf16 xs[MT * DDIM];
  __shared__ __align__(16) __bf16 hs[MT * HCHUNK];
  __shared__ float gw[MT];
  __shared__ int   tok[MT];

  const int e  = blockIdx.x >> 7;
  const int tb = blockIdx.x & 127;
  const int cnt = counts[e];
  if (tb * MT >= cnt) return;

  const int tid = threadIdx.x, lane = tid & 31, wv = tid >> 5;
  const int hi = lane >> 4, ln = lane & 15;

  if (tid < MT) {
    const int slot = tb * MT + tid;
    if (slot < cnt) { tok[tid] = tlist[e * T_TOK + slot]; gw[tid] = wlist[e * T_TOK + slot]; }
    else            { tok[tid] = 0;                        gw[tid] = 0.0f; }
  }
  __syncthreads();
  for (int i = tid; i < MT * (DDIM / 4); i += 256) {
    const int m = i / (DDIM / 4), c4 = i - m * (DDIM / 4);
    const float4 v = ((const float4*)(x + (size_t)tok[m] * DDIM))[c4];
    __bf16* dst = &xs[m * DDIM + c4 * 4];
    dst[0] = (__bf16)v.x; dst[1] = (__bf16)v.y; dst[2] = (__bf16)v.z; dst[3] = (__bf16)v.w;
  }
  __syncthreads();

  const float* w1e = w1 + (size_t)e * HDIM * DDIM;
  const float* w3e = w3 + (size_t)e * HDIM * DDIM;
  const float* w2e = w2 + (size_t)e * DDIM * HDIM;

  for (int half = 0; half < 2; ++half) {
    const int hbase = half * HCHUNK;
    for (int p = 0; p < HCHUNK / 128; ++p) {
      const int ncol = hbase + p * 128 + wv * 16 + ln;
      const float* b1row = w1e + (size_t)ncol * DDIM;
      const float* b3row = w3e + (size_t)ncol * DDIM;
      const __bf16* arow = &xs[ln * DDIM];
      v8f a1 = {}, a3 = {};
      for (int k0 = 0; k0 < DDIM; k0 += 32) {
        const v16bf a  = pack_lds(arow + k0 + hi * 8, arow + k0 + 16 + hi * 8);
        const v16bf b1 = pack_f32(b1row + k0 + hi * 8, b1row + k0 + 16 + hi * 8);
        const v16bf b3 = pack_f32(b3row + k0 + hi * 8, b3row + k0 + 16 + hi * 8);
        a1 = WMMA_BF16(a, b1, a1);
        a3 = WMMA_BF16(a, b3, a3);
      }
      const int ccol = p * 128 + wv * 16 + ln;
#pragma unroll
      for (int r = 0; r < 8; ++r)
        hs[(r + hi * 8) * HCHUNK + ccol] = (__bf16)(silu_fast(a1[r]) * a3[r]);
    }
    __syncthreads();
    for (int q = 0; q < DDIM / 128; ++q) {
      const int dcol = q * 128 + wv * 16 + ln;
      const float* b2row = w2e + (size_t)dcol * HDIM + hbase;
      const __bf16* arow = &hs[ln * HCHUNK];
      v8f ay = {};
      for (int k0 = 0; k0 < HCHUNK; k0 += 32) {
        const v16bf a = pack_lds(arow + k0 + hi * 8, arow + k0 + 16 + hi * 8);
        const v16bf b = pack_f32(b2row + k0 + hi * 8, b2row + k0 + 16 + hi * 8);
        ay = WMMA_BF16(a, b, ay);
      }
#pragma unroll
      for (int r = 0; r < 8; ++r) {
        const int m = r + hi * 8;
        atomicAdd(out + (size_t)tok[m] * DDIM + dcol, gw[m] * ay[r]);
      }
    }
    __syncthreads();
  }
}

// ---------------------------------------------------------------------------
extern "C" void kernel_launch(void* const* d_in, const int* in_sizes, int n_in,
                              void* d_out, int out_size, void* d_ws, size_t ws_size,
                              hipStream_t stream) {
  const float* x      = (const float*)d_in[0];
  const float* gate_w = (const float*)d_in[1];
  const float* w1     = (const float*)d_in[2];
  const float* w2     = (const float*)d_in[3];
  const float* w3     = (const float*)d_in[4];
  float* out = (float*)d_out;

  char* ws = (char*)d_ws;
  int*    counts = (int*)(ws + WS_COUNTS);
  int*    offs   = (int*)(ws + WS_OFFS);
  int*    tlist  = (int*)(ws + WS_TLIST);
  float*  wlist  = (float*)(ws + WS_WLIST);
  __bf16* xg     = (__bf16*)(ws + WS_XG);
  __bf16* hbuf   = (__bf16*)(ws + WS_HBUF);

  moe_init<<<2048, 256, 0, stream>>>(out, counts);
  moe_gate<<<T_TOK / 8, 256, 0, stream>>>(x, gate_w, counts, tlist, wlist);

  if (ws_size >= WS_REQ) {
    moe_offsets<<<1, 1, 0, stream>>>(counts, offs);
    moe_gather<<<EEXP * 256, 256, 0, stream>>>(x, counts, offs, tlist, xg);
    moe_gemm_h<<<EEXP * MBH * NBH, 256, 0, stream>>>(xg, w1, w3, counts, offs, hbuf);
    moe_gemm_y<<<EEXP * MBH * NBY, 256, 0, stream>>>(hbuf, w2, counts, offs, tlist, wlist, out);
  } else {
    moe_ffn_fused<<<EEXP * (T_TOK / MT), 256, 0, stream>>>(x, w1, w2, w3, counts, tlist, wlist, out);
  }
}
